// Trunk_72980084293803
// MI455X (gfx1250) — compile-verified, matrix operand order checked by reading, not run on hardware
//
#include <hip/hip_runtime.h>
#include <math.h>

// ---------------------------------------------------------------------------
// CDNA5 (gfx1250) wave32 WMMA bf16 implementation of the 3-layer transformer.
// fp32 activations in workspace; fp32->bf16 conversion at LDS-staging time so
// every matmul runs on v_wmma_f32_16x16x32_bf16 with fp32 accumulation.
// B operands are staged TRANSPOSED in LDS so both A and B fragments are two
// contiguous 8-element runs -> 2x ds_load_b128 each, no u16 gather/packing.
// ---------------------------------------------------------------------------

typedef __attribute__((ext_vector_type(16))) __bf16 v16bf;
typedef __attribute__((ext_vector_type(8)))  __bf16 v8bf;
typedef __attribute__((ext_vector_type(8)))  float  v8f;

constexpr int kBatch  = 512;
constexpr int kSeq    = 90;
constexpr int kEmb    = 256;
constexpr int kHeads  = 8;
constexpr int kHd     = 32;
constexpr int kFF     = 1024;
constexpr int kLayers = 3;
constexpr int kRows   = kBatch * kSeq;        // 46080 tokens

// fp32 -> bf16 (RNE); native cast lets the backend use v_cvt(_pk)_bf16_f32
__device__ inline __bf16 f2bf(float f) { return (__bf16)f; }

__device__ inline v8f wmma_bf16(v16bf a, v16bf b, v8f c) {
    // D = A(16x32) * B(32x16) + C, fp32 accumulate
    return __builtin_amdgcn_wmma_f32_16x16x32_bf16(
        /*neg_a=*/false, a, /*neg_b=*/false, b,
        /*c_mod=*/(short)0, c, /*reuse_a=*/false, /*reuse_b=*/false);
}

// 16x32 bf16 fragment from LDS, one 32-k row per "lane row", row stride in
// elements. ISA 7.12.2: lane r = lane&15, hi = lane>>4; the lane's 16 values
// are two contiguous runs: k = 8*hi..8*hi+7 and k = 16+8*hi..16+8*hi+7.
// Used for A tiles (row = M) and for transposed-B tiles (row = N).
__device__ inline v16bf load_frag32(const __bf16* lds, int stride, int lane) {
    const int r = lane & 15, hi = lane >> 4;
    const v8bf lo = *(const v8bf*)&lds[r * stride + (hi << 3)];
    const v8bf hh = *(const v8bf*)&lds[r * stride + 16 + (hi << 3)];
    return __builtin_shufflevector(lo, hh, 0, 1, 2, 3, 4, 5, 6, 7,
                                           8, 9, 10, 11, 12, 13, 14, 15);
}

// ---------------------------------------------------------------------------
// out = x + pos_emb[positions]
// ---------------------------------------------------------------------------
__global__ void embed_kernel(const float* __restrict__ x, const int* __restrict__ pos,
                             const float* __restrict__ pemb, float* __restrict__ out) {
    const size_t idx = (size_t)blockIdx.x * blockDim.x + threadIdx.x;   // kRows*kEmb total
    const int    e   = (int)(idx & (kEmb - 1));
    const size_t tok = idx >> 8;                                        // kEmb == 256
    const int    p   = pos[tok];
    out[idx] = x[idx] + pemb[(size_t)p * kEmb + e];
}

// ---------------------------------------------------------------------------
// GEMM: D[M,N] = act(A[M,K] * W[K,N] + bias). Block = BM x BN, 256 threads
// (8 waves), TPW 16x16 C tiles per wave sharing one B fragment.
// Template shapes divide every problem size exactly -> no masking anywhere.
// ---------------------------------------------------------------------------
template <int BM, int BN, int TPW>
__global__ void __launch_bounds__(256)
gemm_kernel(const float* __restrict__ A, const float* __restrict__ W,
            const float* __restrict__ bias, float* __restrict__ D,
            int M, int N, int K, int act) {
    constexpr int TM = BM / 16, TN = BN / 16;
    static_assert(TM * TN == 8 * TPW, "8 waves x TPW tiles");
    constexpr int APT = BM * 32 / 256;   // A elems per thread (contiguous)
    constexpr int BPT = BN / 8;          // B k-elems per thread (one column run)

    __shared__ alignas(16) __bf16 As[BM * 32];   // row-major (M x 32k)
    __shared__ alignas(16) __bf16 Bt[BN * 32];   // TRANSPOSED: (N x 32k)

    const int tid  = threadIdx.x;
    const int lane = tid & 31;
    const int wave = tid >> 5;
    const int nt   = wave % TN;
    const int mt0  = wave / TN;
    const int rowBlock = blockIdx.x * BM;
    const int colBlock = blockIdx.y * BN;

    // A staging: per-thread contiguous run within one row
    const int aE = tid * APT, aR = aE >> 5, aC = aE & 31;
    // B staging: thread owns column bN, k-run bK0..bK0+BPT-1; for each j the
    // 32 lanes of a wave read 32 consecutive floats of one W row (coalesced),
    // and write a contiguous bf16 run of Bt row bN (merges into b64/b128).
    const int bN  = tid % BN;
    const int bK0 = (tid / BN) * BPT;

    v8f acc[TPW];
#pragma unroll
    for (int t = 0; t < TPW; ++t) acc[t] = (v8f){};

    for (int kb = 0; kb < K; kb += 32) {
        const float* aSrc = &A[(size_t)(rowBlock + aR) * K + kb + aC];
        const float* bSrc = &W[(size_t)(kb + bK0) * N + colBlock + bN];
        if (kb + 32 < K) {   // pull next K-block toward L2 (global_prefetch_b8)
            __builtin_prefetch(aSrc + 32, 0, 1);
            __builtin_prefetch(&W[(size_t)(kb + 32 + bK0) * N + colBlock + bN], 0, 1);
        }
#pragma unroll
        for (int j = 0; j < APT; ++j) As[aE + j] = f2bf(aSrc[j]);
#pragma unroll
        for (int j = 0; j < BPT; ++j) Bt[bN * 32 + bK0 + j] = f2bf(bSrc[(size_t)j * N]);
        __syncthreads();

        const v16bf b = load_frag32(&Bt[nt * 16 * 32], 32, lane);
#pragma unroll
        for (int t = 0; t < TPW; ++t) {
            const v16bf a = load_frag32(&As[(mt0 + t * (TM / TPW)) * 16 * 32], 32, lane);
            acc[t] = wmma_bf16(a, b, acc[t]);
        }
        __syncthreads();
    }

    // epilogue: C/D layout -> lane (hi,nl) owns col nl, rows r + 8*hi
    const int nl = lane & 15, hi = lane >> 4;
    const int col = colBlock + nt * 16 + nl;
    const float bv = bias ? bias[col] : 0.f;
#pragma unroll
    for (int t = 0; t < TPW; ++t) {
#pragma unroll
        for (int r = 0; r < 8; ++r) {
            const int row = rowBlock + (mt0 + t * (TM / TPW)) * 16 + (hi << 3) + r;
            float v = acc[t][r] + bv;
            if (act) v = fmaxf(v, 0.f);
            D[(size_t)row * N + col] = v;
        }
    }
}

// ---------------------------------------------------------------------------
// Attention: one workgroup per (n, h). S = (Q K^T)/16, softmax, O = P V.
// 90 padded to 96. qh/kk/vv layout: (n, s, h, d) contiguous (stride 32).
// K is consumed transposed so its fragment is naturally contiguous; V is
// staged transposed (Vt[d][s]) so the O-phase B fragment is contiguous too.
// ---------------------------------------------------------------------------
__global__ void __launch_bounds__(256)
attn_kernel(const float* __restrict__ qh, const float* __restrict__ kk,
            const float* __restrict__ vv, float* __restrict__ o) {
    __shared__ alignas(16) __bf16 Qs[96 * 32];   // (s x d)
    __shared__ alignas(16) __bf16 Ks[96 * 32];   // (s x d)
    __shared__ alignas(16) __bf16 Vt[32 * 96];   // TRANSPOSED: (d x s)
    __shared__ float  Ss[96 * 96];

    const int n    = blockIdx.x >> 3;
    const int h    = blockIdx.x & 7;
    const int tid  = threadIdx.x;
    const int lane = tid & 31;
    const int wave = tid >> 5;

    // Q/K rows (s-major)
    for (int idx = tid; idx < kSeq * 32; idx += 256) {
        const int s = idx >> 5, d = idx & 31;
        const size_t g = ((size_t)(n * kSeq + s) * kHeads + h) * kHd + d;
        Qs[idx] = f2bf(qh[g]);
        Ks[idx] = f2bf(kk[g]);
    }
    for (int idx = kSeq * 32 + tid; idx < 96 * 32; idx += 256) {
        const __bf16 z = f2bf(0.f);
        Qs[idx] = z; Ks[idx] = z;
    }
    // V transposed: thread owns (d, 12-long s-run); lanes are d-consecutive so
    // each global read iteration is 32 consecutive floats (coalesced); LDS
    // writes are a contiguous bf16 run of Vt row d.
    {
        const int d  = tid & 31;
        const int s0 = (tid >> 5) * 12;          // 8 runs of 12 cover 96
#pragma unroll
        for (int j = 0; j < 12; ++j) {
            const int s = s0 + j;
            const float val = (s < kSeq)
                ? vv[((size_t)(n * kSeq + s) * kHeads + h) * kHd + d] : 0.f;
            Vt[d * 96 + s0 + j] = f2bf(val);
        }
    }
    __syncthreads();

    // ---- scores: 6x6 tiles of 16x16, K = 32 (one WMMA each) ----
    for (int t = wave; t < 36; t += 8) {
        const int mt = t / 6, ntt = t % 6;
        const v16bf a = load_frag32(&Qs[mt * 16 * 32], 32, lane);
        // B = K^T tile: lane row = key index -> contiguous d-run of Ks
        const v16bf b = load_frag32(&Ks[ntt * 16 * 32], 32, lane);
        v8f c = {};
        c = wmma_bf16(a, b, c);
        const int nl = lane & 15, hi = lane >> 4;
        const int col = ntt * 16 + nl;
#pragma unroll
        for (int r = 0; r < 8; ++r) {
            const int m = mt * 16 + (hi << 3) + r;
            float sv;
            if (m < kSeq) sv = (col < kSeq) ? c[r] * 0.0625f : -1e30f;  // 1/sqrt(256)
            else          sv = 0.f;
            Ss[m * 96 + col] = sv;
        }
    }
    __syncthreads();

    // ---- softmax over the 90 valid cols; zero the pad cols ----
    if (tid < 96) {
        float* row = &Ss[tid * 96];
        float mx = -1e30f;
        for (int c = 0; c < kSeq; ++c) mx = fmaxf(mx, row[c]);
        float sum = 0.f;
        for (int c = 0; c < kSeq; ++c) { const float e = __expf(row[c] - mx); row[c] = e; sum += e; }
        const float inv = 1.f / sum;
        for (int c = 0; c < kSeq; ++c) row[c] *= inv;
        for (int c = kSeq; c < 96; ++c) row[c] = 0.f;
    }
    __syncthreads();

    // ---- O = P(96x96) * V(96x32): 6x2 tiles, 3 K-steps ----
    for (int t = wave; t < 12; t += 8) {
        const int mt = t >> 1, ntt = t & 1;
        v8f c = {};
        for (int kb = 0; kb < 96; kb += 32) {
            const int ml = lane & 15, hi = lane >> 4;
            v16bf a;   // P fragment: fp32 -> bf16 conversion gather
#pragma unroll
            for (int v = 0; v < 8; ++v) {
                const int k = ((v & 3) << 1) + (hi << 3) + ((v >> 2) << 4);
                a[2 * v]     = f2bf(Ss[(mt * 16 + ml) * 96 + kb + k]);
                a[2 * v + 1] = f2bf(Ss[(mt * 16 + ml) * 96 + kb + k + 1]);
            }
            // V fragment from transposed Vt: contiguous runs of row d
            const v8bf blo = *(const v8bf*)&Vt[(ntt * 16 + ml) * 96 + kb + (hi << 3)];
            const v8bf bhh = *(const v8bf*)&Vt[(ntt * 16 + ml) * 96 + kb + 16 + (hi << 3)];
            const v16bf b = __builtin_shufflevector(blo, bhh, 0, 1, 2, 3, 4, 5, 6, 7,
                                                    8, 9, 10, 11, 12, 13, 14, 15);
            c = wmma_bf16(a, b, c);
        }
        const int nl = lane & 15, hi = lane >> 4;
#pragma unroll
        for (int r = 0; r < 8; ++r) {
            const int m = mt * 16 + (hi << 3) + r;
            if (m < kSeq)
                o[(size_t)(n * kSeq + m) * kEmb + h * kHd + ntt * 16 + nl] = c[r];
        }
    }
}

// ---------------------------------------------------------------------------
// dst = LayerNorm(src + res) * g + b   (one block per token row of 256)
// ---------------------------------------------------------------------------
__global__ void __launch_bounds__(256)
lnres_kernel(const float* __restrict__ src, const float* __restrict__ res,
             const float* __restrict__ g, const float* __restrict__ b,
             float* __restrict__ dst) {
    __shared__ float red[256];
    __shared__ float stat[2];
    const int row = blockIdx.x, t = threadIdx.x;
    const size_t off = (size_t)row * kEmb + t;
    const float v = src[off] + res[off];
    red[t] = v;
    __syncthreads();
    for (int s = 128; s > 0; s >>= 1) { if (t < s) red[t] += red[t + s]; __syncthreads(); }
    if (t == 0) stat[0] = red[0] * (1.f / kEmb);
    __syncthreads();
    const float mean = stat[0];
    const float d = v - mean;
    red[t] = d * d;
    __syncthreads();
    for (int s = 128; s > 0; s >>= 1) { if (t < s) red[t] += red[t + s]; __syncthreads(); }
    if (t == 0) stat[1] = rsqrtf(red[0] * (1.f / kEmb) + 1e-5f);
    __syncthreads();
    dst[off] = d * stat[1] * g[t] + b[t];
}

// ---------------------------------------------------------------------------
// head: sigmoid(tanh(tanh(x*Wc1+b)*Wc2+b)*Wc3+b), one thread per token
// ---------------------------------------------------------------------------
__global__ void head_kernel(const float* __restrict__ xin,
                            const float* __restrict__ Wc1, const float* __restrict__ bc1,
                            const float* __restrict__ Wc2, const float* __restrict__ bc2,
                            const float* __restrict__ Wc3, const float* __restrict__ bc3,
                            float* __restrict__ y) {
    const int row = blockIdx.x * blockDim.x + threadIdx.x;
    if (row >= kRows) return;
    const float* xr = &xin[(size_t)row * kEmb];
    float z1[8];
#pragma unroll
    for (int j = 0; j < 8; ++j) z1[j] = bc1[j];
    for (int e = 0; e < kEmb; ++e) {
        const float xv = xr[e];
#pragma unroll
        for (int j = 0; j < 8; ++j) z1[j] += xv * Wc1[e * 8 + j];
    }
#pragma unroll
    for (int j = 0; j < 8; ++j) z1[j] = tanhf(z1[j]);
    float z2[8];
#pragma unroll
    for (int j = 0; j < 8; ++j) {
        float s = bc2[j];
#pragma unroll
        for (int i = 0; i < 8; ++i) s += z1[i] * Wc2[i * 8 + j];
        z2[j] = tanhf(s);
    }
    float o0 = bc3[0], o1 = bc3[1];
#pragma unroll
    for (int i = 0; i < 8; ++i) { o0 += z2[i] * Wc3[i * 2]; o1 += z2[i] * Wc3[i * 2 + 1]; }
    y[(size_t)row * 2]     = 1.f / (1.f + __expf(-o0));
    y[(size_t)row * 2 + 1] = 1.f / (1.f + __expf(-o1));
}

// ---------------------------------------------------------------------------
extern "C" void kernel_launch(void* const* d_in, const int* in_sizes, int n_in,
                              void* d_out, int out_size, void* d_ws, size_t ws_size,
                              hipStream_t stream) {
    (void)in_sizes; (void)n_in; (void)out_size; (void)ws_size;
    const float* x    = (const float*)d_in[0];
    const float* q    = (const float*)d_in[1];
    const int*   pos  = (const int*)d_in[2];
    const float* pemb = (const float*)d_in[3];
    const float* Wq   = (const float*)d_in[4];
    const float* Wk   = (const float*)d_in[5];
    const float* Wv   = (const float*)d_in[6];
    const float* Wo   = (const float*)d_in[7];
    const float* bo   = (const float*)d_in[8];
    const float* g1   = (const float*)d_in[9];
    const float* b1   = (const float*)d_in[10];
    const float* g2   = (const float*)d_in[11];
    const float* b2   = (const float*)d_in[12];
    const float* Wf1  = (const float*)d_in[13];
    const float* bf1  = (const float*)d_in[14];
    const float* Wf2  = (const float*)d_in[15];
    const float* bf2  = (const float*)d_in[16];
    const float* Wc1  = (const float*)d_in[17];
    const float* bc1  = (const float*)d_in[18];
    const float* Wc2  = (const float*)d_in[19];
    const float* bc2  = (const float*)d_in[20];
    const float* Wc3  = (const float*)d_in[21];
    const float* bc3  = (const float*)d_in[22];
    float* y = (float*)d_out;

    // workspace layout (fp32): out | qh | k | v | attn_o | h | f1(x4)
    const size_t RE = (size_t)kRows * kEmb;
    float* outb = (float*)d_ws;
    float* qhb  = outb + RE;
    float* kbuf = qhb + RE;
    float* vbuf = kbuf + RE;
    float* aob  = vbuf + RE;     // attention output; later reused as FFN2 output
    float* hb   = aob + RE;
    float* f1b  = hb + RE;       // kRows * kFF
    float* aout = qhb;           // Wo output reuses qh (free after attention)
    float* f2b  = aob;           // FFN2 output reuses attn_o (free after Wo GEMM)

    const int M8 = kRows * kHeads;   // 368640 rows of 32 for QKV projections

    embed_kernel<<<kRows, 256, 0, stream>>>(x, pos, pemb, outb);

    for (int i = 0; i < kLayers; ++i) {
        const float* Wqi  = Wq  + (size_t)i * kHd * kHd;
        const float* Wki  = Wk  + (size_t)i * kHd * kHd;
        const float* Wvi  = Wv  + (size_t)i * kHd * kHd;
        const float* Woi  = Wo  + (size_t)i * kEmb * kEmb;
        const float* Wf1i = Wf1 + (size_t)i * kEmb * kFF;
        const float* Wf2i = Wf2 + (size_t)i * kFF * kEmb;

        // per-head-dim QKV projections (queries come from the ORIGINAL q)
        gemm_kernel<64, 32, 1><<<dim3(M8 / 64, 1), 256, 0, stream>>>(
            q, Wqi, nullptr, qhb, M8, kHd, kHd, 0);
        gemm_kernel<64, 32, 1><<<dim3(M8 / 64, 1), 256, 0, stream>>>(
            outb, Wki, nullptr, kbuf, M8, kHd, kHd, 0);
        gemm_kernel<64, 32, 1><<<dim3(M8 / 64, 1), 256, 0, stream>>>(
            outb, Wvi, nullptr, vbuf, M8, kHd, kHd, 0);

        attn_kernel<<<kBatch * kHeads, 256, 0, stream>>>(qhb, kbuf, vbuf, aob);

        // output projection + bias
        gemm_kernel<64, 64, 2><<<dim3(kRows / 64, kEmb / 64), 256, 0, stream>>>(
            aob, Woi, bo + (size_t)i * kEmb, aout, kRows, kEmb, kEmb, 0);

        // h = LN(a + q_original)
        lnres_kernel<<<kRows, 256, 0, stream>>>(aout, q, g1 + (size_t)i * kEmb,
                                                b1 + (size_t)i * kEmb, hb);
        // FFN
        gemm_kernel<64, 64, 2><<<dim3(kRows / 64, kFF / 64), 256, 0, stream>>>(
            hb, Wf1i, bf1 + (size_t)i * kFF, f1b, kRows, kFF, kEmb, 1);
        gemm_kernel<64, 64, 2><<<dim3(kRows / 64, kEmb / 64), 256, 0, stream>>>(
            f1b, Wf2i, bf2 + (size_t)i * kEmb, f2b, kRows, kEmb, kFF, 0);
        // out = LN(f + h)
        lnres_kernel<<<kRows, 256, 0, stream>>>(f2b, hb, g2 + (size_t)i * kEmb,
                                                b2 + (size_t)i * kEmb, outb);
    }

    head_kernel<<<(kRows + 255) / 256, 256, 0, stream>>>(outb, Wc1, bc1, Wc2, bc2, Wc3, bc3, y);
}
